// AverageAttn_62981400428866
// MI455X (gfx1250) — compile-verified, matrix operand order checked by reading, not run on hardware
//
#include <hip/hip_runtime.h>
#include <hip/hip_bf16.h>

typedef __bf16 bf16;
typedef __attribute__((ext_vector_type(16))) __bf16 v16bf;
typedef __attribute__((ext_vector_type(8)))  float  v8f;

#define WMMA_BF16(a, b, c) \
  __builtin_amdgcn_wmma_f32_16x16x32_bf16(false, (a), false, (b), (short)0, (c), false, false)

static constexpr int Bb    = 4;
static constexpr int Ss    = 2048;
static constexpr int Dd    = 1024;
static constexpr int Mrows = Bb * Ss;   // 8192 tokens
static constexpr int LDK   = 48;        // LDS row stride in bf16 for K=32 tiles (96B, 16B-aligned)

// ---------------------------------------------------------------------------
// CDNA5 async global->LDS copy (GLOBAL_LOAD_ASYNC_TO_LDS_B128, ASYNCcnt).
// Param types per hipcc diagnostic: (int4 AS(1)*, int4 AS(3)*, imm, imm).
// Guarded so the kernel still compiles (synchronous fallback) if absent.
// ---------------------------------------------------------------------------
#if __has_builtin(__builtin_amdgcn_global_load_async_to_lds_b128)
#define HAVE_ASYNC_LDS 1
typedef int v4i __attribute__((vector_size(16)));
typedef __attribute__((address_space(1))) v4i gv4i;
typedef __attribute__((address_space(3))) v4i lv4i;
__device__ __forceinline__ void async_cp32B(const bf16* g, bf16* l) {
  // two b128 ops: 16B per lane each; wave-wide this stages 2x512B
  __builtin_amdgcn_global_load_async_to_lds_b128((gv4i*)(void*)g,       (lv4i*)l,       0, 0);
  __builtin_amdgcn_global_load_async_to_lds_b128((gv4i*)(void*)(g + 8), (lv4i*)(l + 8), 0, 0);
}
#if __has_builtin(__builtin_amdgcn_s_wait_asynccnt)
#define WAIT_ASYNC(n) __builtin_amdgcn_s_wait_asynccnt(n)
#else
#define WAIT_ASYNC(n) asm volatile("s_wait_asynccnt %0" ::"i"(n) : "memory")
#endif
#else
#define HAVE_ASYNC_LDS 0
#define WAIT_ASYNC(n) ((void)0)
__device__ __forceinline__ void async_cp32B(const bf16* g, bf16* l) {
  const uint4* s = reinterpret_cast<const uint4*>(g);
  uint4* d = reinterpret_cast<uint4*>(l);
  d[0] = s[0]; d[1] = s[1];
}
#endif

// ---------------------------------------------------------------------------
// Load a 16-element bf16 WMMA fragment (32 bytes) from LDS as two b128 reads.
// ---------------------------------------------------------------------------
__device__ __forceinline__ v16bf ldsFrag(const bf16* p) {
  union { v16bf v; uint4 u[2]; } r;
  const uint4* q = reinterpret_cast<const uint4*>(p);
  r.u[0] = q[0];
  r.u[1] = q[1];
  return r.v;
}

__device__ __forceinline__ float sigmoidf(float x) {
  return 1.0f / (1.0f + __expf(-x));
}

// ---------------------------------------------------------------------------
// Weight transpose + f32->bf16 convert:  Wt[n][k] = (bf16) W[k][n]
// ---------------------------------------------------------------------------
__global__ __launch_bounds__(256)
void wconv(const float* __restrict__ W, bf16* __restrict__ Wt, int Kdim, int Ndim) {
  __shared__ float t[32][33];
  const int nb = blockIdx.x * 32;
  const int kb = blockIdx.y * 32;
  const int tx = threadIdx.x & 31;
  const int ty = threadIdx.x >> 5;           // 0..7
  for (int i = ty; i < 32; i += 8)
    t[i][tx] = W[(size_t)(kb + i) * Ndim + (nb + tx)];
  __syncthreads();
  for (int i = ty; i < 32; i += 8)
    Wt[(size_t)(nb + i) * Kdim + (kb + tx)] = (bf16)t[tx][i];
}

// ---------------------------------------------------------------------------
// Elementwise f32 -> bf16 (for iQ, so gate A-tiles can use async staging).
// ---------------------------------------------------------------------------
__global__ __launch_bounds__(256)
void f32_to_bf16(const float* __restrict__ in, bf16* __restrict__ out) {
  const size_t i = (size_t)blockIdx.x * 256 + threadIdx.x;
  const float4 f = reinterpret_cast<const float4*>(in)[i];
  bf16* d = out + i * 4;
  d[0] = (bf16)f.x; d[1] = (bf16)f.y; d[2] = (bf16)f.z; d[3] = (bf16)f.w;
}

// ---------------------------------------------------------------------------
// Causal cumulative mean of iV along S, written directly as bf16.
// ---------------------------------------------------------------------------
__global__ __launch_bounds__(256)
void avg_scan(const float* __restrict__ iV, bf16* __restrict__ avg) {
  const int d = blockIdx.x * 256 + threadIdx.x;
  const int b = blockIdx.y;
  const float* src = iV + (size_t)b * Ss * Dd + d;
  bf16*        dst = avg + (size_t)b * Ss * Dd + d;
  float sum = 0.0f;
  for (int s = 0; s < Ss; ++s) {
    sum += src[(size_t)s * Dd];
    dst[(size_t)s * Dd] = (bf16)(sum * (1.0f / (float)(s + 1)));
  }
}

// ---------------------------------------------------------------------------
// bf16 WMMA GEMM with async double-buffered LDS staging.
//   C[M,N] = epilogue(A[M,K] * Bt[N,K]^T + bias[N])
// Block tile 128x128, K-step 32, ping-pong buffers. 8 waves (2x4); each wave
// computes 64x32 = 4x2 WMMA tiles. RELU / WF32 / WBF16 select epilogue.
// ---------------------------------------------------------------------------
template <int RELU, int WF32, int WBF16>
__global__ __launch_bounds__(256)
void gemm_bf16(const bf16* __restrict__ A, const bf16* __restrict__ Bt,
               const float* __restrict__ bias,
               float* __restrict__ outf, bf16* __restrict__ outb,
               int M, int N, int K) {
  __shared__ __align__(16) bf16 As[2][128 * LDK];
  __shared__ __align__(16) bf16 Bs[2][128 * LDK];

  const int tid   = threadIdx.x;
  const int lane  = tid & 31;
  const int wid   = tid >> 5;
  const int half  = lane >> 4;
  const int l16   = lane & 15;
  const int waveM = wid >> 2;   // 0..1
  const int waveN = wid & 3;    // 0..3
  const int m0    = blockIdx.y * 128;
  const int n0    = blockIdx.x * 128;
  const int srow  = tid >> 1;   // 0..127
  const int sseg  = tid & 1;    // 0..1

  const bf16* Ag = A  + (size_t)(m0 + srow) * K + sseg * 16;
  const bf16* Bg = Bt + (size_t)(n0 + srow) * K + sseg * 16;
  bf16* Al[2] = { &As[0][srow * LDK + sseg * 16], &As[1][srow * LDK + sseg * 16] };
  bf16* Bl[2] = { &Bs[0][srow * LDK + sseg * 16], &Bs[1][srow * LDK + sseg * 16] };

  auto stage = [&](int k0, int buf) {   // 4 async-b128 per wave per tile
    async_cp32B(Ag + k0, Al[buf]);
    async_cp32B(Bg + k0, Bl[buf]);
  };

  v8f zero = {};
  v8f acc[4][2];
#pragma unroll
  for (int i = 0; i < 4; ++i)
#pragma unroll
    for (int j = 0; j < 2; ++j) acc[i][j] = zero;

  stage(0, 0);
  for (int k0 = 0, it = 0; k0 < K; k0 += 32, ++it) {
    const int buf  = it & 1;
    const bool more = (k0 + 32 < K);
    if (more) stage(k0 + 32, buf ^ 1);         // overlap next tile with compute
    if (more) WAIT_ASYNC(4); else WAIT_ASYNC(0);
    __syncthreads();

    v16bf af[4], bfr[2];
#pragma unroll
    for (int mi = 0; mi < 4; ++mi)
      af[mi] = ldsFrag(&As[buf][(waveM * 64 + mi * 16 + l16) * LDK + half * 16]);
#pragma unroll
    for (int ni = 0; ni < 2; ++ni)
      bfr[ni] = ldsFrag(&Bs[buf][(waveN * 32 + ni * 16 + l16) * LDK + half * 16]);
#pragma unroll
    for (int mi = 0; mi < 4; ++mi)
#pragma unroll
      for (int ni = 0; ni < 2; ++ni)
        acc[mi][ni] = WMMA_BF16(af[mi], bfr[ni], acc[mi][ni]);
    __syncthreads();
  }

  // ---- epilogue: bias (+relu), convert, store ----
#pragma unroll
  for (int ni = 0; ni < 2; ++ni) {
    const int n = n0 + waveN * 32 + ni * 16 + l16;
    const float bv = bias[n];
#pragma unroll
    for (int mi = 0; mi < 4; ++mi) {
      const int mbase = m0 + waveM * 64 + mi * 16 + 8 * half;
#pragma unroll
      for (int r = 0; r < 8; ++r) {
        float v = acc[mi][ni][r] + bv;
        if (RELU) v = fmaxf(v, 0.0f);
        const size_t idx = (size_t)(mbase + r) * N + n;
        if (WF32)  outf[idx] = v;
        if (WBF16) outb[idx] = (bf16)v;
      }
    }
  }
}

// ---------------------------------------------------------------------------
// Fused gate GEMM:  g = sigmoid(concat([iQ,ffn]) @ Wg + bg), out = gi*iQ+gf*ffn
// A-operand comes from pre-converted bf16 iQ / ffn (async-stageable, the
// concat seam never splits a 16-element chunk). Each block computes paired
// output column tiles n (gate-i) and n+1024 (gate-f) -> fully fused epilogue.
// Block tile: M=128, N=64 per half. 8 waves (4x2); wave = 32x32 per half.
// ---------------------------------------------------------------------------
__global__ __launch_bounds__(256)
void gate_fused(const float* __restrict__ iQ, const float* __restrict__ ffn,
                const bf16* __restrict__ iQb, const bf16* __restrict__ ffnb,
                const bf16* __restrict__ Wgt, const float* __restrict__ bg,
                float* __restrict__ out) {
  __shared__ __align__(16) bf16 As[2][128 * LDK];
  __shared__ __align__(16) bf16 BL[2][64 * LDK];
  __shared__ __align__(16) bf16 BR[2][64 * LDK];

  const int tid   = threadIdx.x;
  const int lane  = tid & 31;
  const int wid   = tid >> 5;
  const int half  = lane >> 4;
  const int l16   = lane & 15;
  const int waveM = wid >> 1;   // 0..3
  const int waveN = wid & 1;    // 0..1
  const int m0    = blockIdx.y * 128;
  const int n0    = blockIdx.x * 64;

  const int srow = tid >> 1;          // A stage: 0..127
  const int sseg = tid & 1;
  const int brow = (tid & 127) >> 1;  // B stage: 0..63 within half
  const bool isR = tid >= 128;

  // B source row (upper half of Wg for threads >= 128)
  const int nrow = (isR ? (Dd + n0) : n0) + brow;
  const bf16* Bgp = Wgt + (size_t)nrow * (2 * Dd) + sseg * 16;
  bf16* Bl[2] = { isR ? &BR[0][brow * LDK + sseg * 16] : &BL[0][brow * LDK + sseg * 16],
                  isR ? &BR[1][brow * LDK + sseg * 16] : &BL[1][brow * LDK + sseg * 16] };
  bf16* Al[2] = { &As[0][srow * LDK + sseg * 16], &As[1][srow * LDK + sseg * 16] };

  auto stage = [&](int k0, int buf) {   // 4 async-b128 per wave per tile
    const int kg = k0 + sseg * 16;      // 16-chunk never straddles the concat seam
    const bf16* asrc = (kg < Dd) ? iQb  + (size_t)(m0 + srow) * Dd + kg
                                 : ffnb + (size_t)(m0 + srow) * Dd + (kg - Dd);
    async_cp32B(asrc, Al[buf]);
    async_cp32B(Bgp + k0, Bl[buf]);
  };

  v8f zero = {};
  v8f accL[2][2], accR[2][2];
#pragma unroll
  for (int i = 0; i < 2; ++i)
#pragma unroll
    for (int j = 0; j < 2; ++j) { accL[i][j] = zero; accR[i][j] = zero; }

  stage(0, 0);
  for (int k0 = 0, it = 0; k0 < 2 * Dd; k0 += 32, ++it) {
    const int buf  = it & 1;
    const bool more = (k0 + 32 < 2 * Dd);
    if (more) stage(k0 + 32, buf ^ 1);
    if (more) WAIT_ASYNC(4); else WAIT_ASYNC(0);
    __syncthreads();

    v16bf af[2], bl[2], br[2];
#pragma unroll
    for (int mi = 0; mi < 2; ++mi)
      af[mi] = ldsFrag(&As[buf][(waveM * 32 + mi * 16 + l16) * LDK + half * 16]);
#pragma unroll
    for (int ni = 0; ni < 2; ++ni) {
      bl[ni] = ldsFrag(&BL[buf][(waveN * 32 + ni * 16 + l16) * LDK + half * 16]);
      br[ni] = ldsFrag(&BR[buf][(waveN * 32 + ni * 16 + l16) * LDK + half * 16]);
    }
#pragma unroll
    for (int mi = 0; mi < 2; ++mi)
#pragma unroll
      for (int ni = 0; ni < 2; ++ni) {
        accL[mi][ni] = WMMA_BF16(af[mi], bl[ni], accL[mi][ni]);
        accR[mi][ni] = WMMA_BF16(af[mi], br[ni], accR[mi][ni]);
      }
    __syncthreads();
  }

  // ---- fused sigmoid-gate epilogue (exact f32 iQ/ffn for the combine) ----
#pragma unroll
  for (int ni = 0; ni < 2; ++ni) {
    const int n = n0 + waveN * 32 + ni * 16 + l16;
    const float bLv = bg[n];
    const float bRv = bg[Dd + n];
#pragma unroll
    for (int mi = 0; mi < 2; ++mi) {
      const int mbase = m0 + waveM * 32 + mi * 16 + 8 * half;
#pragma unroll
      for (int r = 0; r < 8; ++r) {
        const size_t idx = (size_t)(mbase + r) * Dd + n;
        const float gi = sigmoidf(accL[mi][ni][r] + bLv);
        const float gf = sigmoidf(accR[mi][ni][r] + bRv);
        out[idx] = gi * iQ[idx] + gf * ffn[idx];
      }
    }
  }
}

// ---------------------------------------------------------------------------
extern "C" void kernel_launch(void* const* d_in, const int* in_sizes, int n_in,
                              void* d_out, int out_size, void* d_ws, size_t ws_size,
                              hipStream_t stream) {
  const float* iQ = (const float*)d_in[0];
  const float* iV = (const float*)d_in[1];
  const float* W1 = (const float*)d_in[2];
  const float* b1 = (const float*)d_in[3];
  const float* W2 = (const float*)d_in[4];
  const float* b2 = (const float*)d_in[5];
  const float* Wg = (const float*)d_in[6];
  const float* bg = (const float*)d_in[7];
  float* out = (float*)d_out;

  // Workspace layout (~108 MB total)
  char* p = (char*)d_ws;
  bf16*  avg  = (bf16*)p;  p += (size_t)Mrows * Dd * sizeof(bf16);          // 16 MB
  bf16*  h    = (bf16*)p;  p += (size_t)Mrows * Dd * sizeof(bf16);          // 16 MB
  float* ffn  = (float*)p; p += (size_t)Mrows * Dd * sizeof(float);         // 32 MB
  bf16*  ffnb = (bf16*)p;  p += (size_t)Mrows * Dd * sizeof(bf16);          // 16 MB
  bf16*  iQb  = (bf16*)p;  p += (size_t)Mrows * Dd * sizeof(bf16);          // 16 MB
  bf16*  W1t  = (bf16*)p;  p += (size_t)Dd * Dd * sizeof(bf16);             //  2 MB
  bf16*  W2t  = (bf16*)p;  p += (size_t)Dd * Dd * sizeof(bf16);             //  2 MB
  bf16*  Wgt  = (bf16*)p;  p += (size_t)(2 * Dd) * (2 * Dd) * sizeof(bf16); //  8 MB

  // 1) weights -> bf16, transposed to N-major for contiguous B-fragments
  wconv<<<dim3(Dd / 32, Dd / 32), 256, 0, stream>>>(W1, W1t, Dd, Dd);
  wconv<<<dim3(Dd / 32, Dd / 32), 256, 0, stream>>>(W2, W2t, Dd, Dd);
  wconv<<<dim3(2 * Dd / 32, 2 * Dd / 32), 256, 0, stream>>>(Wg, Wgt, 2 * Dd, 2 * Dd);

  // 2) iQ -> bf16 (gate A-operand, async-stageable)
  f32_to_bf16<<<dim3((size_t)Mrows * Dd / 4 / 256), 256, 0, stream>>>(iQ, iQb);

  // 3) causal cumulative mean (the tril/einsum is a prefix scan, not a GEMM)
  avg_scan<<<dim3(Dd / 256, Bb), 256, 0, stream>>>(iV, avg);

  // 4) h = relu(avg @ W1 + b1)        -> bf16
  gemm_bf16<1, 0, 1><<<dim3(Dd / 128, Mrows / 128), 256, 0, stream>>>(
      avg, W1t, b1, nullptr, h, Mrows, Dd, Dd);

  // 5) ffn = h @ W2 + b2              -> f32 (epilogue) + bf16 (gate operand)
  gemm_bf16<0, 1, 1><<<dim3(Dd / 128, Mrows / 128), 256, 0, stream>>>(
      h, W2t, b2, ffn, ffnb, Mrows, Dd, Dd);

  // 6) fused gate GEMM + sigmoid + output combine
  gate_fused<<<dim3(Dd / 64, Mrows / 128), 256, 0, stream>>>(
      iQ, ffn, iQb, ffnb, Wgt, bg, out);
}
